// Attentionshare_25709674234614
// MI455X (gfx1250) — matrix operand order repaired and checked by reading, not compiled, not run on hardware
//
#include <hip/hip_runtime.h>
#include <hip/hip_bf16.h>
#include <math.h>

// ---------------------------------------------------------------------------
// Types for CDNA5 WMMA (wave32): v_wmma_f32_16x16x32_bf16
// ---------------------------------------------------------------------------
typedef __attribute__((ext_vector_type(16))) __bf16 v16bf;
typedef __attribute__((ext_vector_type(8)))  __bf16 v8bf;
typedef __attribute__((ext_vector_type(4)))  __bf16 v4bf;
typedef __attribute__((ext_vector_type(8)))  float  v8f;

static __device__ __forceinline__ __bf16 f2bf(float f) {
  unsigned u = __builtin_bit_cast(unsigned, f);
  u += 0x7FFFu + ((u >> 16) & 1u);            // round-to-nearest-even
  unsigned short h = (unsigned short)(u >> 16);
  return __builtin_bit_cast(__bf16, h);
}

#define BM 64
#define BN 64
#define BK 32
#define LDT 40   // LDS row stride (bf16 elems) = 80 bytes -> rows 16B aligned

// ---------------------------------------------------------------------------
// Tile staging. M/N overruns are handled by CLAMPING the index (reads valid
// memory; garbage rows/cols are masked in the epilogue). Only K must be
// zero-padded exactly (it contributes to every output element), so the
// scalar guarded path runs only when the K-tile is partial (K=48 case).
// ---------------------------------------------------------------------------
static __device__ __forceinline__ void
stage_rowmajor(__bf16* __restrict__ dst, const float* __restrict__ src,
               int r0, int k0, int Rmax, int K, int tid, bool fullK)
{
  if (fullK) {
    // 512 float4 loads, branch-free; one ds_store_b64 each
    for (int i = tid; i < (BM * BK) / 4; i += 128) {
      int r  = i >> 3;                 // 0..63
      int c4 = i & 7;                  // 0..7  -> cols c4*4..c4*4+3
      int gr = r0 + r; if (gr > Rmax - 1) gr = Rmax - 1;
      float4 v = *(const float4*)(src + (long)gr * K + k0 + c4 * 4);
      v4bf p;
      p[0] = f2bf(v.x); p[1] = f2bf(v.y); p[2] = f2bf(v.z); p[3] = f2bf(v.w);
      *(v4bf*)&dst[r * LDT + c4 * 4] = p;
    }
  } else {
    for (int i = tid; i < BM * BK; i += 128) {
      int r = i >> 5, c = i & 31;
      int gr = r0 + r; if (gr > Rmax - 1) gr = Rmax - 1;
      int gk = k0 + c;
      float v = (gk < K) ? src[(long)gr * K + gk] : 0.f;
      dst[r * LDT + c] = f2bf(v);
    }
  }
}

// B stored globally as [K,N]; LDS wants B^T [n][k]. Coalesce over N.
static __device__ __forceinline__ void
stage_B_kn(__bf16* __restrict__ dst, const float* __restrict__ src,
           int n0, int k0, int N, int K, int tid, bool fullK)
{
  if (fullK) {
    for (int i = tid; i < (BK * BN) / 4; i += 128) {
      int r  = i >> 4;                 // k: 0..31
      int c4 = i & 15;                 // n group: 0..15
      float4 v = *(const float4*)(src + (long)(k0 + r) * N + n0 + c4 * 4);
      int nb = c4 * 4;
      dst[(nb + 0) * LDT + r] = f2bf(v.x);
      dst[(nb + 1) * LDT + r] = f2bf(v.y);
      dst[(nb + 2) * LDT + r] = f2bf(v.z);
      dst[(nb + 3) * LDT + r] = f2bf(v.w);
    }
  } else {
    for (int i = tid; i < BK * BN; i += 128) {
      int r = i >> 6, c = i & 63;      // r = k, c = n
      int gk = k0 + r;
      float v = (gk < K) ? src[(long)gk * N + n0 + c] : 0.f;
      dst[c * LDT + r] = f2bf(v);
    }
  }
}

// ---------------------------------------------------------------------------
// Generic batched GEMM: C[M,N] = alpha * A[M,K] * B  (opB=0: B[K,N], 1: B[N,K])
// Batch z: offset = (z/P)*s?o + (z%P)*s?i. 128 threads = 4 waves, each wave a
// 32x32 quadrant (4 WMMA accumulators). Double-buffered LDS tiles.
// ---------------------------------------------------------------------------
__global__ void __launch_bounds__(128)
gemm_wmma_bf16(const float* __restrict__ A, const float* __restrict__ B,
               float* __restrict__ C,
               int M, int N, int K, int P,
               long sAo, long sAi, long sBo, long sBi, long sCo, long sCi,
               int opB, float alpha)
{
  __shared__ __bf16 lA[2][BM * LDT];
  __shared__ __bf16 lB[2][BN * LDT];

  const int z = blockIdx.z;
  const float* Ab = A + (long)(z / P) * sAo + (long)(z % P) * sAi;
  const float* Bb = B + (long)(z / P) * sBo + (long)(z % P) * sBi;
  float*       Cb = C + (long)(z / P) * sCo + (long)(z % P) * sCi;

  const int m0 = blockIdx.y * BM;
  const int n0 = blockIdx.x * BN;
  const int tid  = threadIdx.x;
  const int lane = tid & 31;
  const int wid  = tid >> 5;
  const int wr = (wid >> 1) * 32;
  const int wc = (wid & 1) * 32;
  const int l = lane & 15;
  const int g = lane >> 4;

  v8f acc[2][2];
  #pragma unroll
  for (int i = 0; i < 2; ++i)
    #pragma unroll
    for (int j = 0; j < 2; ++j)
      #pragma unroll
      for (int e = 0; e < 8; ++e) acc[i][j][e] = 0.f;

  const int kTiles = (K + BK - 1) / BK;

  // prologue: stage tile 0 into buffer 0
  {
    bool fullK = (BK <= K);
    stage_rowmajor(lA[0], Ab, m0, 0, M, K, tid, fullK);
    if (opB) stage_rowmajor(lB[0], Bb, n0, 0, N, K, tid, fullK);
    else     stage_B_kn   (lB[0], Bb, n0, 0, N, K, tid, fullK);
  }

  for (int kt = 0; kt < kTiles; ++kt) {
    __syncthreads();
    const int cur = kt & 1;

    // ---- issue fragment loads for tile kt (ds_load_b128 x8 per wave) ----
    // A 16x32: lanes 0-15 -> K {0..7,16..23}; lanes 16-31 -> K {8..15,24..31}
    v16bf afrag[2], bfrag[2];
    #pragma unroll
    for (int i = 0; i < 2; ++i) {
      const __bf16* pa = &lA[cur][(wr + 16 * i + l) * LDT + g * 8];
      v8bf alo = *(const v8bf*)pa;             // K = 8g .. 8g+7
      v8bf ahi = *(const v8bf*)(pa + 16);      // K = 16+8g .. 16+8g+7
      #pragma unroll
      for (int e = 0; e < 8; ++e) { afrag[i][e] = alo[e]; afrag[i][8 + e] = ahi[e]; }
    }
    // B 32x16: lanes 0-15 -> K {0..7,16..23}; lanes 16-31 -> K {8..15,24..31}
    #pragma unroll
    for (int j = 0; j < 2; ++j) {
      const __bf16* pb = &lB[cur][(wc + 16 * j + l) * LDT + g * 8];
      v8bf blo = *(const v8bf*)pb;
      v8bf bhi = *(const v8bf*)(pb + 16);
      #pragma unroll
      for (int e = 0; e < 8; ++e) { bfrag[j][e] = blo[e]; bfrag[j][8 + e] = bhi[e]; }
    }

    // ---- stage tile kt+1 into the other buffer (overlaps with WMMA) ----
    if (kt + 1 < kTiles) {
      const int nxt = (kt + 1) & 1;
      const int k0n = (kt + 1) * BK;
      bool fullK = (k0n + BK <= K);
      stage_rowmajor(lA[nxt], Ab, m0, k0n, M, K, tid, fullK);
      if (opB) stage_rowmajor(lB[nxt], Bb, n0, k0n, N, K, tid, fullK);
      else     stage_B_kn   (lB[nxt], Bb, n0, k0n, N, K, tid, fullK);
    }

    #pragma unroll
    for (int i = 0; i < 2; ++i)
      #pragma unroll
      for (int j = 0; j < 2; ++j)
        acc[i][j] = __builtin_amdgcn_wmma_f32_16x16x32_bf16(
            false, afrag[i], false, bfrag[j], (short)0, acc[i][j], false, false);
  }

  // ---- epilogue: C/D layout -> VGPR r: lanes 0-15 M=r, lanes 16-31 M=r+8 ----
  #pragma unroll
  for (int i = 0; i < 2; ++i)
    #pragma unroll
    for (int j = 0; j < 2; ++j) {
      int rowb = m0 + wr + 16 * i + 8 * g;
      int col  = n0 + wc + 16 * j + l;
      if (col < N) {
        #pragma unroll
        for (int r = 0; r < 8; ++r) {
          int rr = rowb + r;
          if (rr < M) Cb[(long)rr * N + col] = alpha * acc[i][j][r];
        }
      }
    }
}

// ---------------------------------------------------------------------------
// Depthwise 3x3, stride 1, zero ("SAME") padding.
// ---------------------------------------------------------------------------
__global__ void __launch_bounds__(256)
dwconv3x3(const float* __restrict__ in, const float* __restrict__ w,
          float* __restrict__ out, long total, int Ctot, int H, int W)
{
  long idx = (long)blockIdx.x * blockDim.x + threadIdx.x;
  if (idx >= total) return;
  int x = (int)(idx % W);
  long t = idx / W;
  int y = (int)(t % H);
  long nc = t / H;
  int c = (int)(nc % Ctot);
  const float* wp = w + (long)c * 9;
  const float* ip = in + nc * (long)H * W;
  float s = 0.f;
  #pragma unroll
  for (int dy = -1; dy <= 1; ++dy) {
    int yy = y + dy;
    if (yy < 0 || yy >= H) continue;
    #pragma unroll
    for (int dx = -1; dx <= 1; ++dx) {
      int xx = x + dx;
      if (xx < 0 || xx >= W) continue;
      s += ip[(long)yy * W + xx] * wp[(dy + 1) * 3 + (dx + 1)];
    }
  }
  out[idx] = s;
}

// ---------------------------------------------------------------------------
// Per-row 1/max(||row||2, 1e-12) over S elems. blockIdx.x = n*rows + c.
// ---------------------------------------------------------------------------
__global__ void __launch_bounds__(256)
row_invnorm(const float* __restrict__ qkv, float* __restrict__ invn,
            int S, long nstride, int rowsPerImage)
{
  int b = blockIdx.x;
  int n = b / rowsPerImage;
  int c = b % rowsPerImage;
  const float* p = qkv + (long)n * nstride + (long)c * S;
  float ss = 0.f;
  for (int i = threadIdx.x; i < S; i += 256) { float v = p[i]; ss += v * v; }
  __shared__ float red[256];
  red[threadIdx.x] = ss;
  __syncthreads();
  for (int s = 128; s > 0; s >>= 1) {
    if (threadIdx.x < s) red[threadIdx.x] += red[threadIdx.x + s];
    __syncthreads();
  }
  if (threadIdx.x == 0) {
    float nm = fmaxf(sqrtf(red[0]), 1e-12f);
    invn[b] = 1.f / nm;
  }
}

// ---------------------------------------------------------------------------
// logits[c,d] = G[c,d]*invq[c]*invk[d]*temp4[h]; softmax over d (48 wide).
// grid: (16 = n*8+h, 48 = c), block: 64.
// ---------------------------------------------------------------------------
__global__ void __launch_bounds__(64)
gram_softmax(float* __restrict__ G, const float* __restrict__ invn,
             const float* __restrict__ temp4)
{
  int zz = blockIdx.x;              // n*8 + h
  int c  = blockIdx.y;
  int n  = zz >> 3, h = zz & 7;
  float* row = G + ((long)zz * 48 + c) * 48;
  const float* iq = invn + (long)n * 768 + h * 48;         // q rows
  const float* ik = invn + (long)n * 768 + 384 + h * 48;   // k rows
  float t = temp4[h];
  int d = threadIdx.x;
  bool act = (d < 48);
  __shared__ float sv[64];
  float x = 0.f;
  if (act) x = row[d] * iq[c] * ik[d] * t;
  sv[d] = act ? x : -3.402823e38f;
  __syncthreads();
  for (int s = 32; s > 0; s >>= 1) {
    if (d < s) sv[d] = fmaxf(sv[d], sv[d + s]);
    __syncthreads();
  }
  float mx = sv[0];
  __syncthreads();
  float e = act ? expf(x - mx) : 0.f;
  sv[d] = e;
  __syncthreads();
  for (int s = 32; s > 0; s >>= 1) {
    if (d < s) sv[d] += sv[d + s];
    __syncthreads();
  }
  if (act) row[d] = e / sv[0];
}

// ---------------------------------------------------------------------------
// Host-side orchestration
// ---------------------------------------------------------------------------
extern "C" void kernel_launch(void* const* d_in, const int* in_sizes, int n_in,
                              void* d_out, int out_size, void* d_ws, size_t ws_size,
                              hipStream_t stream) {
  (void)in_sizes; (void)n_in; (void)out_size; (void)ws_size;
  const float* x      = (const float*)d_in[0];
  const float* qkv_w  = (const float*)d_in[2];
  const float* dw_w   = (const float*)d_in[3];
  const float* proj_w = (const float*)d_in[4];
  const float* temp4  = (const float*)d_in[7];
  float* out = (float*)d_out;

  const int  N = 2, C = 384, H = 128, W = 128;
  const int  S  = H * W;            // 16384
  const int  C3 = 3 * C;            // 1152
  const long qkvElems = (long)N * C3 * S;

  float* ws       = (float*)d_ws;
  float* qkv_pre  = ws;                               // N*1152*S
  float* qkv_dw   = qkv_pre + qkvElems;               // N*1152*S
  float* outc     = qkv_dw + qkvElems;                // N*384*S
  float* invn     = outc + (long)N * C * S;           // N*768
  float* attn     = invn + (long)N * 768;             // 16*48*48

  // 1) qkv = W_qkv[1152x384] @ X[n][384x16384]
  {
    dim3 g(S / BN, C3 / BM, N), b(128);
    gemm_wmma_bf16<<<g, b, 0, stream>>>(
        qkv_w, x, qkv_pre, C3, S, C, /*P=*/1,
        0, 0, (long)C * S, 0, (long)C3 * S, 0, /*opB=*/0, 1.f);
  }
  // 2) depthwise 3x3 (SAME)
  {
    long total = qkvElems;
    long blocks = (total + 255) / 256;
    dwconv3x3<<<dim3((unsigned)blocks), dim3(256), 0, stream>>>(
        qkv_pre, dw_w, qkv_dw, total, C3, H, W);
  }
  // 3) 1/||row|| for q (ch 0..383) and k (ch 384..767)
  row_invnorm<<<dim3(N * 768), dim3(256), 0, stream>>>(
      qkv_dw, invn, S, (long)C3 * S, 768);
  // 4) Gram: G[nh][48x48] = q[nh] @ k[nh]^T  (K=16384, opB=1)
  {
    dim3 g(1, 1, 16), b(128);
    gemm_wmma_bf16<<<g, b, 0, stream>>>(
        qkv_dw, qkv_dw + (long)C * S, attn, 48, 48, S, /*P=*/8,
        (long)C3 * S, 48L * S, (long)C3 * S, 48L * S,
        8L * 48 * 48, 48L * 48, /*opB=*/1, 1.f);
  }
  // 5) scale by norms & temp4, softmax over 48
  gram_softmax<<<dim3(16, 48), dim3(64), 0, stream>>>(attn, invn, temp4);
  // 6) out_c[nh][48x16384] = attn[48x48] @ v[nh][48x16384]
  {
    dim3 g(S / BN, 1, 16), b(128);
    gemm_wmma_bf16<<<g, b, 0, stream>>>(
        attn, qkv_dw + 2L * C * S, outc, 48, S, 48, /*P=*/8,
        8L * 48 * 48, 48L * 48, (long)C3 * S, 48L * S,
        (long)C * S, 48L * S, /*opB=*/0, 1.f);
  }
  // 7) out = 2 * (W_proj[384x384] @ out_c[n][384x16384])   (folds out+out)
  {
    dim3 g(S / BN, C / BM, N), b(128);
    gemm_wmma_bf16<<<g, b, 0, stream>>>(
        proj_w, outc, out, C, S, C, /*P=*/1,
        0, 0, (long)C * S, 0, (long)C * S, 0, /*opB=*/0, 2.f);
  }
}
